// GRU_11338713661455
// MI455X (gfx1250) — matrix operand order repaired and checked by reading, not compiled, not run on hardware
//
#include <hip/hip_runtime.h>
#include <stdint.h>

// ---------------- types ----------------
typedef __bf16  v16bf __attribute__((ext_vector_type(16)));
typedef float   v8f   __attribute__((ext_vector_type(8)));
typedef unsigned u32x4 __attribute__((ext_vector_type(4)));

union Bf16Vec { u32x4 q[2]; v16bf v; };

#define B_  32
#define T_  2048
#define I_  512
#define H_  512
#define BH_ (B_ * H_)          // 16384
#define RB_ 16                 // batch rows per recurrence WG (one WMMA M-tile)

// ---------------- helpers ----------------
__device__ __forceinline__ unsigned short f32_to_bf16(float f) {
  unsigned u = __builtin_bit_cast(unsigned, f);
  u += 0x7FFFu + ((u >> 16) & 1u);       // round-to-nearest-even
  return (unsigned short)(u >> 16);
}

__device__ __forceinline__ float sigmoid_fast(float x) {
  return 1.0f / (1.0f + __expf(-x));
}
__device__ __forceinline__ float tanh_fast(float x) {
  float e = __expf(2.0f * x);
  return (e - 1.0f) / (e + 1.0f);
}

__device__ __forceinline__ v8f vzero8() {
  return (v8f){0.f,0.f,0.f,0.f,0.f,0.f,0.f,0.f};
}

// A-operand (16x32 bf16, row-major [rows][ldk]); works for global or LDS pointers.
// lane<16: row=lane, K=kb+0..15 ; lane>=16: row=lane-16, K=kb+16..31
__device__ __forceinline__ v16bf load_a_tile(const unsigned short* A,
                                             int lane, int rowBase, int ldk, int kb) {
  int row = rowBase + (lane & 15);
  int k0  = kb + ((lane & 16) ? 16 : 0);
  const u32x4* p0 = reinterpret_cast<const u32x4*>(A + (size_t)row * ldk + k0);
  const u32x4* p1 = reinterpret_cast<const u32x4*>(A + (size_t)row * ldk + k0 + 8);
  Bf16Vec u; u.q[0] = *p0; u.q[1] = *p1;
  return u.v;
}

// B-operand (32x16 bf16) from transposed weights BT[N][ldk]:
// lane<16: col=lane, K=kb+0..15 ; lane>=16: col=lane-16, K=kb+16..31
__device__ __forceinline__ v16bf load_bT_tile(const unsigned short* BT,
                                              int lane, int colBase, int ldk, int kb) {
  int col = colBase + (lane & 15);
  int k0  = kb + ((lane & 16) ? 16 : 0);
  const u32x4* p = reinterpret_cast<const u32x4*>(BT + (size_t)col * ldk + k0);
  Bf16Vec u; u.q[0] = p[0]; u.q[1] = p[1];
  return u.v;
}

// async copy of one 16B chunk: Global -> LDS (CDNA5 async path, tracked by ASYNCcnt)
__device__ __forceinline__ void async_g2l_b128(unsigned lds_byte_addr, const void* gp) {
  asm volatile("global_load_async_to_lds_b128 %0, %1, off"
               :: "v"(lds_byte_addr), "v"((unsigned long long)(size_t)gp)
               : "memory");
}
__device__ __forceinline__ void wait_async0() {
  asm volatile("s_wait_asynccnt 0x0" ::: "memory");
}

// ---------------- prep kernels ----------------
__global__ __launch_bounds__(256)
void cvt_bf16(const float* __restrict__ src, unsigned short* __restrict__ dst, size_t n) {
  size_t stride = (size_t)gridDim.x * blockDim.x;
  for (size_t i = (size_t)blockIdx.x * blockDim.x + threadIdx.x; i < n; i += stride)
    dst[i] = f32_to_bf16(src[i]);
}

// src [G][K][N] -> dst [G][N][K] (bf16)
__global__ __launch_bounds__(256)
void cvt_transpose(const float* __restrict__ src, unsigned short* __restrict__ dst,
                   int G, int K, int N) {
  size_t total = (size_t)G * K * N;
  size_t stride = (size_t)gridDim.x * blockDim.x;
  for (size_t i = (size_t)blockIdx.x * blockDim.x + threadIdx.x; i < total; i += stride) {
    int g = (int)(i / ((size_t)K * N));
    int r = (int)(i % ((size_t)K * N));
    int k = r / N, n = r % N;
    dst[((size_t)g * N + n) * K + k] = f32_to_bf16(src[i]);
  }
}

// ---------------- phase 1: gx = x @ W_x + bias ----------------
// grid (512, 8, 3), block 256; wave w handles mtile = bx*8+w, a 16x64 N-strip.
__global__ __launch_bounds__(256)
void gemm_gx(const unsigned short* __restrict__ xbf,     // [B*T][I] bf16
             const unsigned short* __restrict__ wxT,     // [3][H][I] bf16 transposed
             const float* __restrict__ bias,             // [3][H]
             float* __restrict__ gx) {                   // [T][3][B][H]
  const int lane  = threadIdx.x & 31;
  const int w     = threadIdx.x >> 5;
  const int mtile = blockIdx.x * 8 + w;
  const int nBase = blockIdx.y * 64;
  const int g     = blockIdx.z;
  const unsigned short* wT = wxT + (size_t)g * H_ * I_;
  const int mrow = mtile * 16;

  v8f acc[4];
#pragma unroll
  for (int nt = 0; nt < 4; ++nt) acc[nt] = vzero8();

  for (int kb = 0; kb < I_; kb += 32) {
    v16bf a = load_a_tile(xbf, lane, mrow, I_, kb);
#pragma unroll
    for (int nt = 0; nt < 4; ++nt) {
      v16bf b = load_bT_tile(wT, lane, nBase + nt * 16, I_, kb);
      acc[nt] = __builtin_amdgcn_wmma_f32_16x16x32_bf16(
          false, a, false, b, (short)0, acc[nt], false, false);
    }
  }

  const int rlo = (lane & 16) ? 8 : 0;
#pragma unroll
  for (int nt = 0; nt < 4; ++nt) {
    int col = nBase + nt * 16 + (lane & 15);
    float bv = bias[g * H_ + col];
#pragma unroll
    for (int j = 0; j < 8; ++j) {
      int m  = mrow + rlo + j;
      int bb = m >> 11;            // m / T_
      int tt = m & (T_ - 1);       // m % T_
      gx[(((size_t)tt * 3 + g) * B_ + bb) * H_ + col] = acc[nt][j] + bv;
    }
  }
}

// ---------------- phase 2: recurrence, batch-split, LDS-resident ----------------
// 2 WGs x 1024 threads. WG owns batch rows [wg*16, wg*16+16). All state in LDS:
//   [0)      h_bf16  [16][512]  (A-operand of z/r GEMMs)       16 KB
//   [16384)  rh_bf16 [16][512]  (A-operand of candidate GEMM)  16 KB
//   [32768)  h_f32   [16][512]                                 32 KB
//   [65536)  gx slabs x2: [3][16][512] f32 (async dbl-buffer)  192 KB
__global__ __launch_bounds__(1024, 1)
void gru_recurrence(const unsigned short* __restrict__ whT,  // [3][H][H] bf16 transposed
                    const float* __restrict__ gx,            // [T][3][B][H]
                    float* __restrict__ out) {               // [B][T][H] + [B][H]
  extern __shared__ char smem[];
  unsigned short* hbf  = (unsigned short*)smem;
  unsigned short* rhbf = (unsigned short*)(smem + 16384);
  float*          hff  = (float*)(smem + 32768);
  float*          slab0 = (float*)(smem + 65536);
  float*          slab1 = (float*)(smem + 65536 + 98304);

  const int tid   = threadIdx.x;
  const int lane  = tid & 31;
  const int w     = tid >> 5;            // wave 0..31
  const int rBase = blockIdx.x * RB_;    // batch-row base (0 or 16)
  const int p     = w & 15;              // column-pair index (32 cols each)
  const int ga    = w >> 4;              // phase-A gate: 0 = z, 1 = r
  const int n0    = p * 32;
  const int rlo   = (lane & 16) ? 8 : 0;

  // zero LDS state
  for (int i = tid; i < RB_ * H_; i += 1024) { hbf[i] = 0; hff[i] = 0.0f; }

  // prologue: async-copy gx slab for t=0 (6144 x 16B)
  for (int idx = tid; idx < 6144; idx += 1024) {
    int g = idx >> 11, r = (idx >> 7) & 15, c = idx & 127;
    const float* gp = gx + (size_t)g * BH_ + (size_t)(rBase + r) * H_ + c * 4;
    async_g2l_b128(65536u + (unsigned)idx * 16u, gp);
  }
  wait_async0();
  __syncthreads();

  float zreg[2][8];
#pragma unroll
  for (int nt = 0; nt < 2; ++nt)
#pragma unroll
    for (int j = 0; j < 8; ++j) zreg[nt][j] = 0.0f;

  for (int t = 0; t < T_; ++t) {
    const float* gxc = (t & 1) ? slab1 : slab0;

    // ---- phase A: z (waves 0-15) / r (waves 16-31): h_prev @ W_h[g], 2 N-tiles/wave
    {
      const unsigned short* bt = whT + (size_t)ga * H_ * H_;
      v8f c0 = vzero8(), c1 = vzero8();
      for (int kb = 0; kb < H_; kb += 32) {
        v16bf a  = load_a_tile(hbf, lane, 0, H_, kb);               // LDS (ds_load_b128)
        v16bf b0 = load_bT_tile(bt, lane, n0,      H_, kb);         // L2-resident weights
        v16bf b1 = load_bT_tile(bt, lane, n0 + 16, H_, kb);
        c0 = __builtin_amdgcn_wmma_f32_16x16x32_bf16(false, a, false, b0, (short)0, c0, false, false);
        c1 = __builtin_amdgcn_wmma_f32_16x16x32_bf16(false, a, false, b1, (short)0, c1, false, false);
      }
      const float* gxg = gxc + ga * RB_ * H_;
      if (ga == 0) {
#pragma unroll
        for (int nt = 0; nt < 2; ++nt)
#pragma unroll
          for (int j = 0; j < 8; ++j) {
            int row = rlo + j, col = n0 + nt * 16 + (lane & 15);
            float v = nt ? c1[j] : c0[j];
            zreg[nt][j] = sigmoid_fast(gxg[row * H_ + col] + v);
          }
      } else {
#pragma unroll
        for (int nt = 0; nt < 2; ++nt)
#pragma unroll
          for (int j = 0; j < 8; ++j) {
            int row = rlo + j, col = n0 + nt * 16 + (lane & 15);
            float v  = nt ? c1[j] : c0[j];
            float r  = sigmoid_fast(gxg[row * H_ + col] + v);
            float rh = r * hff[row * H_ + col];
            rhbf[row * H_ + col] = f32_to_bf16(rh);
          }
      }
    }
    __syncthreads();

    // ---- phase B: waves 0-15: h_tilde GEMM + blend; waves 16-31: async prefetch t+1
    if (w < 16) {
      const unsigned short* bt = whT + (size_t)2 * H_ * H_;
      v8f c0 = vzero8(), c1 = vzero8();
      for (int kb = 0; kb < H_; kb += 32) {
        v16bf a  = load_a_tile(rhbf, lane, 0, H_, kb);              // LDS
        v16bf b0 = load_bT_tile(bt, lane, n0,      H_, kb);
        v16bf b1 = load_bT_tile(bt, lane, n0 + 16, H_, kb);
        c0 = __builtin_amdgcn_wmma_f32_16x16x32_bf16(false, a, false, b0, (short)0, c0, false, false);
        c1 = __builtin_amdgcn_wmma_f32_16x16x32_bf16(false, a, false, b1, (short)0, c1, false, false);
      }
      const float* gxh = gxc + 2 * RB_ * H_;
#pragma unroll
      for (int nt = 0; nt < 2; ++nt)
#pragma unroll
        for (int j = 0; j < 8; ++j) {
          int row = rlo + j, col = n0 + nt * 16 + (lane & 15);
          float v  = nt ? c1[j] : c0[j];
          float ht = tanh_fast(gxh[row * H_ + col] + v);
          float hp = hff[row * H_ + col];
          float z  = zreg[nt][j];
          float hn = (1.0f - z) * hp + z * ht;
          hff[row * H_ + col] = hn;
          hbf[row * H_ + col] = f32_to_bf16(hn);
          out[((size_t)(rBase + row) * T_ + t) * H_ + col] = hn;
        }
    } else {
      if (t + 1 < T_) {
        unsigned sbase = 65536u + (unsigned)((t + 1) & 1) * 98304u;
        const float* g1 = gx + (size_t)(t + 1) * 3 * BH_;
        for (int idx = tid - 512; idx < 6144; idx += 512) {
          int g = idx >> 11, r = (idx >> 7) & 15, c = idx & 127;
          const float* gp = g1 + (size_t)g * BH_ + (size_t)(rBase + r) * H_ + c * 4;
          async_g2l_b128(sbase + (unsigned)idx * 16u, gp);
        }
      }
      wait_async0();   // copy overlaps the candidate GEMM of waves 0-15
    }
    __syncthreads();
  }

  // h_last tail
  float* tail = out + (size_t)B_ * T_ * H_ + (size_t)rBase * H_;
  for (int i = tid; i < RB_ * H_; i += 1024) tail[i] = hff[i];
}

// ---------------- host entry ----------------
extern "C" void kernel_launch(void* const* d_in, const int* in_sizes, int n_in,
                              void* d_out, int out_size, void* d_ws, size_t ws_size,
                              hipStream_t stream) {
  const float* x    = (const float*)d_in[0];   // [B,T,I]
  const float* Wx   = (const float*)d_in[1];   // [3,I,H]
  const float* Wh   = (const float*)d_in[2];   // [3,H,H]
  const float* bias = (const float*)d_in[3];   // [3,H]
  float* out = (float*)d_out;

  // workspace: x_bf16 | WxT | WhT | gx
  char* ws = (char*)d_ws;
  unsigned short* xbf = (unsigned short*)ws;                  // B*T*I
  unsigned short* wxT = xbf + (size_t)B_ * T_ * I_;           // 3*H*I
  unsigned short* whT = wxT + (size_t)3 * H_ * I_;            // 3*H*H
  float*          gx  = (float*)(whT + (size_t)3 * H_ * H_);  // T*3*B*H

  cvt_bf16<<<2048, 256, 0, stream>>>(x, xbf, (size_t)B_ * T_ * I_);
  cvt_transpose<<<512, 256, 0, stream>>>(Wx, wxT, 3, I_, H_);
  cvt_transpose<<<512, 256, 0, stream>>>(Wh, whT, 3, H_, H_);

  dim3 g1(512, 8, 3);
  gemm_gx<<<g1, 256, 0, stream>>>(xbf, wxT, bias, gx);

  // 2 persistent WGs, 256 KB dynamic LDS each (state + double-buffered gx slabs)
  gru_recurrence<<<2, 1024, 262144, stream>>>(whT, gx, out);
}
